// LSTMnet_28552942583877
// MI455X (gfx1250) — matrix operand order repaired and checked by reading, not compiled, not run on hardware
//
#include <hip/hip_runtime.h>
#include <math.h>

// Problem sizes (match reference)
#define SEQT 2048
#define EMB  512
#define HID  512
#define G4   2048   // 4*HID
#define VOC  32000

#define SCAN_WGS 64           // workgroups cooperating on the recurrence
#define UNITS_PER_WG (HID / SCAN_WGS)   // 8 hidden units per WG
#define ROWS_PER_WG (4 * UNITS_PER_WG)  // 32 W_hh rows per WG

typedef __attribute__((ext_vector_type(16))) __bf16 v16bf;
typedef __attribute__((ext_vector_type(2)))  __bf16 bf16x2;
typedef __attribute__((ext_vector_type(8)))  float  v8f;

// ---------------------------------------------------------------------------
// WMMA fragment packing per CDNA5 ISA 7.12.2 (wave32), lane l: hf=l>>4, mn=l&15
//   A v16bf: e in [0,8)  -> A[mn][k0 + 8*hf + e]
//            e in [8,16) -> A[mn][k0 + 16 + 8*hf + (e-8)]
//   B v16bf: same k order as A -> W[col mn][k]   (B = W^T, W row-major)
//   C/D v8f: reg r       -> C[r + 8*hf][mn]
// ---------------------------------------------------------------------------
__device__ __forceinline__ v16bf pack_a(const float* __restrict__ arow, int k0, int hf) {
  v16bf a;
#pragma unroll
  for (int e = 0; e < 8; ++e) {
    a[e]     = (__bf16)arow[k0 + 8 * hf + e];
    a[e + 8] = (__bf16)arow[k0 + 16 + 8 * hf + e];
  }
  return a;
}

__device__ __forceinline__ v16bf pack_b(const float* __restrict__ brow, int k0, int hf) {
  v16bf b;
#pragma unroll
  for (int e = 0; e < 8; ++e) {
    b[e]     = (__bf16)brow[k0 + 8 * hf + e];
    b[e + 8] = (__bf16)brow[k0 + 16 + 8 * hf + e];
  }
  return b;
}

// One wave computes a 32x32 output block as 2x2 register-blocked 16x16 tiles:
// per K-step, 4 fragment loads feed 4 WMMAs (2x the arithmetic intensity of 1x1).
__device__ __forceinline__ void gemm2x2(const float* __restrict__ a0,
                                        const float* __restrict__ a1,
                                        const float* __restrict__ b0,
                                        const float* __restrict__ b1,
                                        int K, int hf, v8f acc[4]) {
#pragma unroll 4
  for (int k0 = 0; k0 < K; k0 += 32) {
    v16bf fa0 = pack_a(a0, k0, hf);
    v16bf fa1 = pack_a(a1, k0, hf);
    v16bf fb0 = pack_b(b0, k0, hf);
    v16bf fb1 = pack_b(b1, k0, hf);
    acc[0] = __builtin_amdgcn_wmma_f32_16x16x32_bf16(false, fa0, false, fb0, (short)0, acc[0], false, false);
    acc[1] = __builtin_amdgcn_wmma_f32_16x16x32_bf16(false, fa0, false, fb1, (short)0, acc[1], false, false);
    acc[2] = __builtin_amdgcn_wmma_f32_16x16x32_bf16(false, fa1, false, fb0, (short)0, acc[2], false, false);
    acc[3] = __builtin_amdgcn_wmma_f32_16x16x32_bf16(false, fa1, false, fb1, (short)0, acc[3], false, false);
  }
}

// ---------------------------------------------------------------------------
// Kernel 1: x_gates[t][n] = sum_k embed[seq[t]][k] * W_ih[n][k] + b_ih[n] + b_hh[n]
// One wave per 32x32 block; embedding gather fused into the A-row pointers.
// ---------------------------------------------------------------------------
__global__ void lstm_xgates_kernel(const int* __restrict__ seq,
                                   const float* __restrict__ embed,
                                   const float* __restrict__ W_ih,
                                   const float* __restrict__ b_ih,
                                   const float* __restrict__ b_hh,
                                   float* __restrict__ xg) {
  const int gw   = blockIdx.x * (blockDim.x >> 5) + (threadIdx.x >> 5);
  const int lane = threadIdx.x & 31;
  const int tilesN = G4 / 32;                 // 64
  const int tm = gw / tilesN;
  const int tn = gw % tilesN;
  if (tm >= SEQT / 32) return;

  const int hf = lane >> 4;
  const int mn = lane & 15;

  const float* a0 = embed + (long)seq[tm * 32 + mn] * EMB;
  const float* a1 = embed + (long)seq[tm * 32 + 16 + mn] * EMB;
  const int n0 = tn * 32 + mn;
  const int n1 = n0 + 16;
  const float* b0 = W_ih + (long)n0 * EMB;
  const float* b1 = W_ih + (long)n1 * EMB;

  const float bias0 = b_ih[n0] + b_hh[n0];
  const float bias1 = b_ih[n1] + b_hh[n1];
  v8f acc[4];
#pragma unroll
  for (int r = 0; r < 8; ++r) {
    acc[0][r] = bias0; acc[1][r] = bias1;
    acc[2][r] = bias0; acc[3][r] = bias1;
  }

  gemm2x2(a0, a1, b0, b1, EMB, hf, acc);

#pragma unroll
  for (int mi = 0; mi < 2; ++mi)
#pragma unroll
    for (int ni = 0; ni < 2; ++ni)
#pragma unroll
      for (int r = 0; r < 8; ++r)
        xg[(long)(tm * 32 + mi * 16 + r + 8 * hf) * G4 + tn * 32 + ni * 16 + mn]
            = acc[mi * 2 + ni][r];
}

// ---------------------------------------------------------------------------
// Zero the grid-sync counter (d_ws is poisoned; must re-zero every launch).
// ---------------------------------------------------------------------------
__global__ void lstm_init_kernel(unsigned* __restrict__ cnt) {
  if (threadIdx.x == 0 && blockIdx.x == 0) *cnt = 0u;
}

// ---------------------------------------------------------------------------
// Kernel 2: cooperative LSTM scan across SCAN_WGS workgroups.
// Each WG owns 8 hidden units -> 32 rows of W_hh, held bf16 in LDS (32 KB),
// so the per-step matvec streams from LDS instead of re-reading 4 MB from L2.
// h is published per step through a double-buffered global buffer; the grid
// step-syncs with a release/acquire atomic counter (64 co-resident WGs).
// ---------------------------------------------------------------------------
__global__ void lstm_scan_kernel(const float* __restrict__ xg,
                                 const float* __restrict__ W_hh,
                                 float* __restrict__ hs,
                                 float* __restrict__ hbuf,   // 2 x HID floats
                                 unsigned* __restrict__ cnt) {
  __shared__ __bf16 Wsl[ROWS_PER_WG * HID];   // 32 rows x 512 bf16 = 32 KB
  __shared__ float  h_sh[HID];
  __shared__ float  gate_sh[ROWS_PER_WG];

  const int w   = blockIdx.x;                 // 0..SCAN_WGS-1
  const int tid = threadIdx.x;                // 0..255

  // Stage this WG's W_hh slice into LDS (bf16). Local row r:
  //   gate = r / UNITS_PER_WG, unit u = r % UNITS_PER_WG,
  //   global row = gate*HID + w*UNITS_PER_WG + u.
  for (int idx = tid; idx < ROWS_PER_WG * HID; idx += 256) {
    const int r = idx >> 9;                   // idx / HID
    const int k = idx & (HID - 1);
    const int grow = (r / UNITS_PER_WG) * HID + w * UNITS_PER_WG + (r % UNITS_PER_WG);
    Wsl[idx] = (__bf16)W_hh[(long)grow * HID + k];
  }
  for (int i = tid; i < HID; i += 256) h_sh[i] = 0.0f;
  __syncthreads();

  const int row = tid >> 3;                   // 0..31: local W_hh row
  const int seg = tid & 7;                    // 8-way K split per row
  const int j   = w * UNITS_PER_WG + (tid & (UNITS_PER_WG - 1)); // unit (tid<8)
  float c = 0.0f;

  for (int t = 0; t < SEQT; ++t) {
    // --- partial dot: 64-long K segment of <W row, h> from LDS ---
    float p = 0.0f;
    const bf16x2* __restrict__ wr2 = (const bf16x2*)&Wsl[row * HID + seg * 64];
    const float2* __restrict__ hr2 = (const float2*)&h_sh[seg * 64];
#pragma unroll 8
    for (int k = 0; k < 32; ++k) {
      const bf16x2 wv = wr2[k];
      const float2 hv = hr2[k];
      p += (float)wv.x * hv.x + (float)wv.y * hv.y;
    }
    // reduce the 8 segments (consecutive lanes) with wave shuffles
    p += __shfl_xor(p, 1);
    p += __shfl_xor(p, 2);
    p += __shfl_xor(p, 4);
    if (seg == 0) gate_sh[row] = p;
    __syncthreads();

    // --- c/h update for this WG's 8 units ---
    if (tid < UNITS_PER_WG) {
      const float* __restrict__ xgt = xg + (long)t * G4;
      const float gi = gate_sh[tid]                     + xgt[0 * HID + j];
      const float gf = gate_sh[UNITS_PER_WG + tid]      + xgt[1 * HID + j];
      const float gg = gate_sh[2 * UNITS_PER_WG + tid]  + xgt[2 * HID + j];
      const float go = gate_sh[3 * UNITS_PER_WG + tid]  + xgt[3 * HID + j];
      const float i_ = 1.0f / (1.0f + __expf(-gi));
      const float f_ = 1.0f / (1.0f + __expf(-gf));
      const float g_ = 1.0f - 2.0f / (__expf(2.0f * gg) + 1.0f);   // tanh
      const float o_ = 1.0f / (1.0f + __expf(-go));
      c = f_ * c + i_ * g_;
      const float hj = o_ * (1.0f - 2.0f / (__expf(2.0f * c) + 1.0f));
      hbuf[((t + 1) & 1) * HID + j] = hj;
      hs[(long)t * HID + j] = hj;
      __threadfence();                        // make h slice visible pre-arrive
    }
    __syncthreads();

    // --- grid step barrier: release arrive + acquire spin ---
    if (tid == 0) {
      __hip_atomic_fetch_add(cnt, 1u, __ATOMIC_RELEASE, __HIP_MEMORY_SCOPE_AGENT);
      const unsigned target = (unsigned)(t + 1) * (unsigned)SCAN_WGS;
      while (__hip_atomic_load(cnt, __ATOMIC_ACQUIRE, __HIP_MEMORY_SCOPE_AGENT) < target)
        __builtin_amdgcn_s_sleep(1);
    }
    __syncthreads();
    __threadfence();                          // discard stale h before reload

    // --- reload full h for next step ---
    const float* __restrict__ hbr = hbuf + ((t + 1) & 1) * HID;
    for (int i = tid; i < HID; i += 256) h_sh[i] = hbr[i];
    __syncthreads();
  }
}

// ---------------------------------------------------------------------------
// Kernel 3: logits[t][v] = sum_k hs[t][k] * W_out[v][k] + b_out[v]
// One wave per 32x32 block; 64 x 1000 blocks. Writes straight into d_out.
// ---------------------------------------------------------------------------
__global__ void lstm_logits_kernel(const float* __restrict__ hs,
                                   const float* __restrict__ W_out,
                                   const float* __restrict__ b_out,
                                   float* __restrict__ out) {
  const int gw   = blockIdx.x * (blockDim.x >> 5) + (threadIdx.x >> 5);
  const int lane = threadIdx.x & 31;
  const int tilesN = VOC / 32;                // 1000
  const int tm = gw / tilesN;
  const int tn = gw % tilesN;
  if (tm >= SEQT / 32) return;

  const int hf = lane >> 4;
  const int mn = lane & 15;

  const float* a0 = hs + (long)(tm * 32 + mn) * HID;
  const float* a1 = hs + (long)(tm * 32 + 16 + mn) * HID;
  const int n0 = tn * 32 + mn;
  const int n1 = n0 + 16;
  const float* b0 = W_out + (long)n0 * HID;
  const float* b1 = W_out + (long)n1 * HID;

  const float bias0 = b_out[n0];
  const float bias1 = b_out[n1];
  v8f acc[4];
#pragma unroll
  for (int r = 0; r < 8; ++r) {
    acc[0][r] = bias0; acc[1][r] = bias1;
    acc[2][r] = bias0; acc[3][r] = bias1;
  }

  gemm2x2(a0, a1, b0, b1, HID, hf, acc);

#pragma unroll
  for (int mi = 0; mi < 2; ++mi)
#pragma unroll
    for (int ni = 0; ni < 2; ++ni)
#pragma unroll
      for (int r = 0; r < 8; ++r)
        out[(long)(tm * 32 + mi * 16 + r + 8 * hf) * VOC + tn * 32 + ni * 16 + mn]
            = acc[mi * 2 + ni][r];
}

// ---------------------------------------------------------------------------
// Kernel 4: in-place log_softmax over each 32000-wide row of d_out.
// ---------------------------------------------------------------------------
__global__ void lstm_logsoftmax_kernel(float* __restrict__ logits) {
  const int row = blockIdx.x;
  float* __restrict__ x = logits + (long)row * VOC;
  __shared__ float sm[256];
  const int tid = threadIdx.x;

  float m = -INFINITY;
  for (int i = tid; i < VOC; i += 256) m = fmaxf(m, x[i]);
  sm[tid] = m;
  __syncthreads();
  for (int s = 128; s > 0; s >>= 1) {
    if (tid < s) sm[tid] = fmaxf(sm[tid], sm[tid + s]);
    __syncthreads();
  }
  m = sm[0];
  __syncthreads();

  float sum = 0.0f;
  for (int i = tid; i < VOC; i += 256) sum += __expf(x[i] - m);
  sm[tid] = sum;
  __syncthreads();
  for (int s = 128; s > 0; s >>= 1) {
    if (tid < s) sm[tid] += sm[tid + s];
    __syncthreads();
  }
  const float lse = m + __logf(sm[0]);

  for (int i = tid; i < VOC; i += 256) x[i] = x[i] - lse;
}

// ---------------------------------------------------------------------------
extern "C" void kernel_launch(void* const* d_in, const int* in_sizes, int n_in,
                              void* d_out, int out_size, void* d_ws, size_t ws_size,
                              hipStream_t stream) {
  (void)in_sizes; (void)n_in; (void)out_size; (void)ws_size;

  const int*   seq   = (const int*)d_in[0];
  const float* embed = (const float*)d_in[1];
  const float* W_ih  = (const float*)d_in[2];
  const float* W_hh  = (const float*)d_in[3];
  const float* b_ih  = (const float*)d_in[4];
  const float* b_hh  = (const float*)d_in[5];
  const float* W_out = (const float*)d_in[6];
  const float* b_out = (const float*)d_in[7];
  float* out = (float*)d_out;

  // workspace: x_gates (16 MB) | hs (4 MB) | hbuf (2*HID f32) | counter
  float*    xg   = (float*)d_ws;
  float*    hs   = xg + (size_t)SEQT * G4;
  float*    hbuf = hs + (size_t)SEQT * HID;
  unsigned* cnt  = (unsigned*)(hbuf + 2 * HID);

  // 1) embedding + input projection: 64x64 blocks of 32x32, 8 waves/block
  {
    const int waves = (SEQT / 32) * (G4 / 32);      // 4096
    lstm_xgates_kernel<<<waves / 8, 256, 0, stream>>>(seq, embed, W_ih, b_ih, b_hh, xg);
  }
  // 2) reset sync counter, then cooperative scan across 64 WGs
  lstm_init_kernel<<<1, 64, 0, stream>>>(cnt);
  lstm_scan_kernel<<<SCAN_WGS, 256, 0, stream>>>(xg, W_hh, hs, hbuf, cnt);
  // 3) output projection: 64x1000 blocks of 32x32, 8 waves/block
  {
    const int waves = (SEQT / 32) * (VOC / 32);     // 64000
    lstm_logits_kernel<<<waves / 8, 256, 0, stream>>>(hs, W_out, b_out, out);
  }
  // 4) in-place log_softmax per row
  lstm_logsoftmax_kernel<<<SEQT, 256, 0, stream>>>(out);
}
